// DualRecModel_44152263802933
// MI455X (gfx1250) — matrix operand order repaired and checked by reading, not compiled, not run on hardware
//
#include <hip/hip_runtime.h>
#include <math.h>

// ---------------------------------------------------------------------------
// Model constants (match reference)
// ---------------------------------------------------------------------------
#define SD      512     // D
#define NH      8       // N_HEAD
#define DH      64      // D_HEAD
#define DINNER  2048
#define SEG     512
#define NLAYER  2
#define BATCH   32
#define ROWS    (SEG * BATCH)      // 16384 token rows, row index r = s*BATCH + b
#define HSTRIDE (BATCH * NH * DH)  // 16384 floats between consecutive s in (S,B,N,D)

typedef __attribute__((ext_vector_type(2)))  float    v2f;
typedef __attribute__((ext_vector_type(8)))  float    v8f;
typedef __attribute__((ext_vector_type(16))) __bf16   v16bf;
typedef __attribute__((ext_vector_type(4)))  unsigned v4u;
typedef __attribute__((ext_vector_type(4)))  int      v4i;
typedef __attribute__((ext_vector_type(8)))  int      v8i;

// fp32 WMMA: D(16x16) = A(16x4) * B(4x16) + C
__device__ __forceinline__ v8f wmma4(v2f a, v2f b, v8f c) {
  return __builtin_amdgcn_wmma_f32_16x16x4_f32(
      false, a, false, b, (short)0, c, false, false);
}

// bf16 WMMA: D(16x16 f32) = A(16x32 bf16) * B(32x16 bf16) + C
union Frag16 { unsigned u[8]; v16bf v; };
__device__ __forceinline__ v8f wmma32bf(const Frag16& a, const Frag16& b, v8f c) {
  return __builtin_amdgcn_wmma_f32_16x16x32_bf16(
      false, a.v, false, b.v, (short)0, c, false, false);
}

// f32 -> bf16 (round-to-nearest-even), packed pair
__device__ __forceinline__ unsigned f2bf_rne(float x) {
  unsigned u = __float_as_uint(x);
  return (u + 0x7FFFu + ((u >> 16) & 1u)) >> 16;
}
__device__ __forceinline__ unsigned pack2(float lo, float hi) {
  return f2bf_rne(lo) | (f2bf_rne(hi) << 16);
}

// ---------------------------------------------------------------------------
// Tensor Data Mover: 2D tile load (f32 elements) into LDS.
// tile_w elements contiguous per line, tile_h lines, row_stride elements apart.
// D# layout per CDNA5 ISA (2D: groups 2/3 zero).
// ---------------------------------------------------------------------------
__device__ __forceinline__ void tdm_load_2d_f32(
    const float* gptr, unsigned lds_off, int tile_w, int tile_h, long row_stride)
{
  unsigned long long ga = (unsigned long long)(size_t)gptr;   // byte address
  v4u g0;
  g0[0] = 1u;                                                 // count=1 (valid user D#)
  g0[1] = lds_off;                                            // lds_addr  [63:32]
  g0[2] = (unsigned)(ga & 0xFFFFFFFFu);                       // global_addr lo
  g0[3] = (unsigned)((ga >> 32) & 0x1FFFFFFu) | (2u << 30);   // global_addr hi | type=2
  unsigned tw = (unsigned)tile_w, thh = (unsigned)tile_h;
  unsigned long long st0 = (unsigned long long)row_stride;    // in data_size units
  v8i g1;
  g1[0] = (int)(2u << 16);                                    // data_size=2 (4 bytes)
  g1[1] = (int)(tw << 16);                                    // tensor_dim0[15:0] @ [63:48]
  g1[2] = (int)((tw >> 16) | (thh << 16));                    // tdim0 hi | tensor_dim1 lo
  g1[3] = (int)((thh >> 16) | (tw << 16));                    // tdim1 hi | tile_dim0
  g1[4] = (int)(thh & 0xFFFFu);                               // tile_dim1 (tile_dim2=0)
  g1[5] = (int)(unsigned)(st0 & 0xFFFFFFFFu);                 // tensor_dim0_stride lo
  g1[6] = (int)(unsigned)((st0 >> 32) & 0xFFFFu);             // stride hi (stride1=0)
  g1[7] = 0;
  v4i z4 = {0, 0, 0, 0};
#if __has_include(<hip/amd_detail/amd_gfx1250_TDM.h>)
  v8i z8 = {0, 0, 0, 0, 0, 0, 0, 0};
  __builtin_amdgcn_tensor_load_to_lds(g0, g1, z4, z4, z8, 0);
#else
  __builtin_amdgcn_tensor_load_to_lds(g0, g1, z4, z4, 0);
#endif
}

// ---------------------------------------------------------------------------
// Embedding gather: h[s*B+b, :] = item_emb[input_ids[b, s], :]
// ---------------------------------------------------------------------------
__global__ __launch_bounds__(256) void gather_kernel(
    const int* __restrict__ ids, const float* __restrict__ emb,
    float* __restrict__ h)
{
  int r = blockIdx.x;            // r = s*BATCH + b
  int s = r / BATCH;
  int b = r - s * BATCH;
  long tok = ids[(long)b * SEG + s];
  const float* src = emb + tok * SD;
  float* dst = h + (long)r * SD;
  for (int d = threadIdx.x; d < SD; d += 256) dst[d] = src[d];
}

// ---------------------------------------------------------------------------
// Sinusoidal relative position embedding: pos[(2S) x D]
// ---------------------------------------------------------------------------
__global__ __launch_bounds__(256) void posemb_kernel(float* __restrict__ pos)
{
  int p = blockIdx.x;            // 0..1023
  float seq = (float)(SEG - p);
  int k = threadIdx.x;           // one freq per thread
  float invf = expf(-((float)(2 * k) / (float)SD) * logf(10000.0f));
  float ang = seq * invf;
  pos[(long)p * SD + k]       = sinf(ang);
  pos[(long)p * SD + 256 + k] = cosf(ang);
}

// ---------------------------------------------------------------------------
// bf16 WMMA GEMM with TDM-staged tiles:
//   C[M,N] = A[M,K] * B[K,N] (+bias)(+GELU) ; A row-major (lda=sAm),
//   B row-major K x N (bT=0, ldB between k-rows) or transposed N x K (bT=1,
//   ldB between n-rows; contiguous axis is k).
// Block: 256 threads = 8 waves; tile 64(M) x 32(N); K chunk 32.
// Wave 0 issues two tensor_load_to_lds per chunk, s_wait_tensorcnt, then all
// threads pack f32 -> bf16 fragments laid out per the 16-bit WMMA VGPR map.
// epi: 0 none, 1 +bias, 2 +bias,+exact GELU
// ---------------------------------------------------------------------------
__global__ __launch_bounds__(256) void gemm_bf16_wmma_kernel(
    const float* __restrict__ A, long sAm,
    const float* __restrict__ Bm, long ldB, int bT,
    float* __restrict__ C, long sCm,
    int K, const float* __restrict__ bias, int epi)
{
  __shared__ float    Ast[64 * 32];   // TDM dest: A tile, row-major 64x32
  __shared__ float    Bst[32 * 32];   // TDM dest: B tile ([k][n] or [n][k])
  __shared__ unsigned Abf[64][17];    // packed bf16 K-pairs, [m][pair]
  __shared__ unsigned Bbf[16][33];    // packed bf16 K-pairs, [pair][n]

  const int tid  = threadIdx.x;
  const int wave = tid >> 5;
  const int lane = tid & 31;
  const int half = lane >> 4;
  const int l16  = lane & 15;
  const int m0   = blockIdx.y * 64;
  const int n0   = blockIdx.x * 32;
  const int wm   = wave & 3;
  const int wn   = wave >> 2;

  const unsigned ldsA = (unsigned)(size_t)Ast;
  const unsigned ldsB = (unsigned)(size_t)Bst;

  v8f acc = {};
  for (int k0 = 0; k0 < K; k0 += 32) {
    __syncthreads();                              // prev chunk fully consumed
    if (wave == 0) {
      tdm_load_2d_f32(A + (long)m0 * sAm + k0, ldsA, 32, 64, sAm);
      if (!bT)  // B[k][n] tile: 32 lines of 32 contiguous n
        tdm_load_2d_f32(Bm + (long)k0 * ldB + n0, ldsB, 32, 32, ldB);
      else      // B^T: 32 lines (n) of 32 contiguous k
        tdm_load_2d_f32(Bm + (long)n0 * ldB + k0, ldsB, 32, 32, ldB);
      __builtin_amdgcn_s_wait_tensorcnt((short)0);
    }
    __syncthreads();                              // tiles landed in LDS

    // pack A: 64 rows x 16 K-pairs
    for (int idx = tid; idx < 64 * 16; idx += 256) {
      int r = idx >> 4, p = idx & 15;
      Abf[r][p] = pack2(Ast[r * 32 + 2 * p], Ast[r * 32 + 2 * p + 1]);
    }
    // pack B: 16 K-pairs x 32 cols (transpose during pack when bT)
    for (int idx = tid; idx < 16 * 32; idx += 256) {
      int p = idx >> 5, n = idx & 31;
      float lo = bT ? Bst[n * 32 + 2 * p]     : Bst[(2 * p) * 32 + n];
      float hi = bT ? Bst[n * 32 + 2 * p + 1] : Bst[(2 * p + 1) * 32 + n];
      Bbf[p][n] = pack2(lo, hi);
    }
    __syncthreads();                              // bf16 fragments ready

    // Fragment gather per 16-bit WMMA VGPR layout:
    //  A VGPR v<4 : K = 2v + 8*half      -> pair v + 4*half
    //  A VGPR v>=4: K = 16+2(v-4)+8*half -> pair 8 + (v-4) + 4*half
    //  B VGPR v   : same K as A VGPR v   -> pair pa
    Frag16 fa, fb;
    const int arow = wm * 16 + l16;
    const int bcol = wn * 16 + l16;
#pragma unroll
    for (int v = 0; v < 8; ++v) {
      int pa = (v < 4) ? (v + 4 * half) : (8 + (v - 4) + 4 * half);
      fa.u[v] = Abf[arow][pa];
      fb.u[v] = Bbf[pa][bcol];
    }
    acc = wmma32bf(fa, fb, acc);
  }

  const int col = n0 + wn * 16 + l16;
  const float bv = (epi > 0) ? bias[col] : 0.0f;
  for (int vv = 0; vv < 8; ++vv) {
    int mrow = m0 + wm * 16 + vv + 8 * half;      // C: VGPR v -> M = v + 8*half
    float x = acc[vv] + bv;
    if (epi == 2) x = 0.5f * x * (1.0f + erff(x * 0.70710678118654752f));
    C[(long)mrow * sCm + col] = x;
  }
}

// ---------------------------------------------------------------------------
// Attention (fp32 WMMA) for one 16-row i-tile of one (b, head):
//   score = (q+rwb)·k^T + relshift((q+rrb)·k_r^T) ; mask; softmax; prob·V
// relshift identity: bd[i,j] = bd_raw[i, j + SEG - i]
// ---------------------------------------------------------------------------
__global__ __launch_bounds__(256) void attn_kernel(
    const float* __restrict__ q,  const float* __restrict__ k,
    const float* __restrict__ v,  const float* __restrict__ kr,
    const float* __restrict__ imask,
    const float* __restrict__ rwb, const float* __restrict__ rrb,
    float* __restrict__ av)
{
  __shared__ float score[16][516];
  __shared__ float qw[16][68];
  __shared__ float qr[16][68];
  __shared__ float pv[2][16][68];
  __shared__ float red[16][17];

  const int tid  = threadIdx.x;
  const int wave = tid >> 5;
  const int lane = tid & 31;
  const int half = lane >> 4;
  const int l16  = lane & 15;
  const int i0   = blockIdx.x * 16;
  const int bn   = blockIdx.y;
  const int b    = bn / NH;
  const int n    = bn - b * NH;
  const int OMG[NH] = {2, 3, 4, 5, 7, 11, 21, 50};
  const int omg = OMG[n];

  for (int idx = tid; idx < 16 * DH; idx += 256) {
    int r = idx >> 6, d = idx & 63;
    float qv = q[(long)(i0 + r) * HSTRIDE + (long)b * (NH * DH) + n * DH + d];
    qw[r][d] = qv + rwb[n * DH + d];
    qr[r][d] = qv + rrb[n * DH + d];
  }
  __syncthreads();

  // ac: 32 j-tiles split across 8 waves
  const float* kbase = k + (long)b * (NH * DH) + n * DH;
  for (int jt = wave; jt < 32; jt += 8) {
    v8f acc = {};
    long jrow = (long)(jt * 16 + l16) * HSTRIDE;
    for (int kk = 0; kk < DH; kk += 4) {
      v2f a, bf;
      a.x = qw[l16][kk + 2 * half + 0];
      a.y = qw[l16][kk + 2 * half + 1];
      bf.x = kbase[jrow + kk + 2 * half + 0];
      bf.y = kbase[jrow + kk + 2 * half + 1];
      acc = wmma4(a, bf, acc);
    }
    for (int vv = 0; vv < 8; ++vv)
      score[vv + 8 * half][jt * 16 + l16] = acc[vv];
  }
  __syncthreads();

  // bd tiles, scatter-add with rel-shift. jj needed: [497-i0, 1023-i0]
  int t0 = 497 - i0; if (t0 < 0) t0 = 0; t0 >>= 4;
  int t1 = (1023 - i0) >> 4;
  const float* krb = kr + n * DH;
  for (int t = t0 + wave; t <= t1; t += 8) {
    v8f acc = {};
    long jjrow = (long)(t * 16 + l16) * (NH * DH);
    for (int kk = 0; kk < DH; kk += 4) {
      v2f a, bf;
      a.x = qr[l16][kk + 2 * half + 0];
      a.y = qr[l16][kk + 2 * half + 1];
      bf.x = krb[jjrow + kk + 2 * half + 0];
      bf.y = krb[jjrow + kk + 2 * half + 1];
      acc = wmma4(a, bf, acc);
    }
    for (int vv = 0; vv < 8; ++vv) {
      int ml = vv + 8 * half;
      int i  = i0 + ml;
      int j  = (t * 16 + l16) + i - SEG;   // inverse rel-shift
      if (j >= 0 && j < SEG) score[ml][j] += acc[vv];
    }
  }
  __syncthreads();

  // mask + softmax (16 threads per row)
  {
    const int row  = tid >> 4;
    const int tcol = tid & 15;
    const int irow = i0 + row;
    const float mi = imask[(long)b * SEG + irow];
    float lmax = -3.4e38f;
    for (int j = tcol; j < SEG; j += 16) {
      float dm = mi + imask[(long)b * SEG + j] + ((j > irow) ? 1.0f : 0.0f);
      int diff = irow - j;
      bool msok = (diff < omg) && (diff > -omg);
      float pen = ((dm + (msok ? 0.0f : 1.0f)) > 0.0f) ? 1e30f : 0.0f;
      float sc = score[row][j] * 0.125f - pen;
      score[row][j] = sc;
      lmax = fmaxf(lmax, sc);
    }
    red[row][tcol] = lmax;
    __syncthreads();
    float rmax = -3.4e38f;
    for (int t = 0; t < 16; ++t) rmax = fmaxf(rmax, red[row][t]);
    __syncthreads();
    float lsum = 0.0f;
    for (int j = tcol; j < SEG; j += 16) {
      float e = expf(score[row][j] - rmax);
      score[row][j] = e;
      lsum += e;
    }
    red[row][tcol] = lsum;
    __syncthreads();
    float rsum = 0.0f;
    for (int t = 0; t < 16; ++t) rsum += red[row][t];
    float inv = 1.0f / rsum;
    for (int j = tcol; j < SEG; j += 16) score[row][j] *= inv;
  }
  __syncthreads();

  // attn_vec: prob(16x512) @ V(512x64); split-K across 2 wave groups
  {
    const int dt = wave & 3;
    const int wg = wave >> 2;
    v8f acc = {};
    const float* vbase = v + (long)b * (NH * DH) + n * DH + dt * 16 + l16;
    for (int kk = 256 * wg; kk < 256 * wg + 256; kk += 4) {
      v2f a, bf;
      a.x = score[l16][kk + 2 * half + 0];
      a.y = score[l16][kk + 2 * half + 1];
      bf.x = vbase[(long)(kk + 2 * half + 0) * HSTRIDE];
      bf.y = vbase[(long)(kk + 2 * half + 1) * HSTRIDE];
      acc = wmma4(a, bf, acc);
    }
    for (int vv = 0; vv < 8; ++vv)
      pv[wg][vv + 8 * half][dt * 16 + l16] = acc[vv];
  }
  __syncthreads();
  for (int idx = tid; idx < 16 * DH; idx += 256) {
    int r = idx >> 6, d = idx & 63;
    av[(long)(i0 + r) * HSTRIDE + (long)b * (NH * DH) + n * DH + d] =
        pv[0][r][d] + pv[1][r][d];
  }
}

// ---------------------------------------------------------------------------
// out = LayerNorm(x + res) * w + b   (row length 512; out may alias res)
// ---------------------------------------------------------------------------
__global__ __launch_bounds__(128) void add_ln_kernel(
    const float* __restrict__ x, const float* __restrict__ res,
    const float* __restrict__ w, const float* __restrict__ bb,
    float* __restrict__ out)
{
  __shared__ float s1[128], s2[128];
  const int row = blockIdx.x, tid = threadIdx.x;
  const float* xr = x + (long)row * SD;
  const float* rr = res + (long)row * SD;
  float vals[4]; float sum = 0.f, sq = 0.f;
  for (int i = 0; i < 4; ++i) {
    float t = xr[tid + 128 * i] + rr[tid + 128 * i];
    vals[i] = t; sum += t; sq += t * t;
  }
  s1[tid] = sum; s2[tid] = sq;
  __syncthreads();
  for (int st = 64; st > 0; st >>= 1) {
    if (tid < st) { s1[tid] += s1[tid + st]; s2[tid] += s2[tid + st]; }
    __syncthreads();
  }
  float mu  = s1[0] * (1.0f / SD);
  float var = s2[0] * (1.0f / SD) - mu * mu;
  float inv = rsqrtf(var + 1e-8f);
  for (int i = 0; i < 4; ++i) {
    int c = tid + 128 * i;
    out[(long)row * SD + c] = (vals[i] - mu) * inv * w[c] + bb[c];
  }
}

// out[b, s, :] = h[s*BATCH+b, :]
__global__ __launch_bounds__(256) void transpose_out_kernel(
    const float* __restrict__ h, float* __restrict__ out)
{
  int r = blockIdx.x;           // r = b*SEG + s
  int b = r >> 9, s = r & 511;
  const float* src = h + (long)(s * BATCH + b) * SD;
  float* dst = out + (long)r * SD;
  for (int d = threadIdx.x; d < SD; d += 256) dst[d] = src[d];
}

// ---------------------------------------------------------------------------
// Host orchestration
// ---------------------------------------------------------------------------
extern "C" void kernel_launch(void* const* d_in, const int* in_sizes, int n_in,
                              void* d_out, int out_size, void* d_ws, size_t ws_size,
                              hipStream_t stream) {
  (void)in_sizes; (void)n_in; (void)out_size; (void)ws_size;
  const int*   input_ids  = (const int*)  d_in[0];
  const float* input_mask = (const float*)d_in[1];
  const float* item_emb   = (const float*)d_in[2];
  const float* Wq         = (const float*)d_in[3];
  const float* Wk         = (const float*)d_in[4];
  const float* Wv         = (const float*)d_in[5];
  const float* Wr         = (const float*)d_in[6];
  const float* Wo         = (const float*)d_in[7];
  const float* r_r_bias   = (const float*)d_in[8];
  const float* r_w_bias   = (const float*)d_in[9];
  const float* ln_attn_w  = (const float*)d_in[10];
  const float* ln_attn_b  = (const float*)d_in[11];
  const float* W1         = (const float*)d_in[12];
  const float* b1         = (const float*)d_in[13];
  const float* W2         = (const float*)d_in[14];
  const float* b2         = (const float*)d_in[15];
  const float* ln_ff_w    = (const float*)d_in[16];
  const float* ln_ff_b    = (const float*)d_in[17];

  float* ws = (float*)d_ws;
  size_t off = 0;
  auto alloc = [&](size_t nf) { float* p = ws + off; off += nf; return p; };
  float* h    = alloc((size_t)ROWS * SD);
  float* qb   = alloc((size_t)ROWS * SD);
  float* kb   = alloc((size_t)ROWS * SD);
  float* vb   = alloc((size_t)ROWS * SD);
  float* av   = alloc((size_t)ROWS * SD);
  float* ao   = alloc((size_t)ROWS * SD);
  float* pos  = alloc((size_t)2 * SEG * SD);
  float* kr   = alloc((size_t)2 * SEG * SD);
  float* t1   = alloc((size_t)2048 * DINNER);
  float* t2   = alloc((size_t)2048 * SD);

  auto gemm = [&](const float* A, long sAm, const float* Bp, long ldB, int bT,
                  float* Cp, long sCm, int M, int N, int K,
                  const float* bias, int epi) {
    dim3 g(N / 32, M / 64);
    gemm_bf16_wmma_kernel<<<g, 256, 0, stream>>>(A, sAm, Bp, ldB, bT,
                                                 Cp, sCm, K, bias, epi);
  };

  gather_kernel<<<ROWS, 256, 0, stream>>>(input_ids, item_emb, h);
  posemb_kernel<<<2 * SEG, 256, 0, stream>>>(pos);

  for (int l = 0; l < NLAYER; ++l) {
    const float* Wq_l = Wq + (size_t)l * SD * SD;
    const float* Wk_l = Wk + (size_t)l * SD * SD;
    const float* Wv_l = Wv + (size_t)l * SD * SD;
    const float* Wr_l = Wr + (size_t)l * SD * SD;
    const float* Wo_l = Wo + (size_t)l * SD * SD;
    const float* rrb  = r_r_bias + (size_t)l * NH * DH;
    const float* rwb  = r_w_bias + (size_t)l * NH * DH;
    const float* W1_l = W1 + (size_t)l * SD * DINNER;
    const float* b1_l = b1 + (size_t)l * DINNER;
    const float* W2_l = W2 + (size_t)l * DINNER * SD;
    const float* b2_l = b2 + (size_t)l * SD;

    // q/k/v/r projections (bf16 WMMA, TDM-staged)
    gemm(h,   SD, Wq_l, SD, 0, qb, SD, ROWS,    SD, SD, nullptr, 0);
    gemm(h,   SD, Wk_l, SD, 0, kb, SD, ROWS,    SD, SD, nullptr, 0);
    gemm(h,   SD, Wv_l, SD, 0, vb, SD, ROWS,    SD, SD, nullptr, 0);
    gemm(pos, SD, Wr_l, SD, 0, kr, SD, 2 * SEG, SD, SD, nullptr, 0);

    // attention (fp32 WMMA)
    attn_kernel<<<dim3(SEG / 16, BATCH * NH), 256, 0, stream>>>(
        qb, kb, vb, kr, input_mask, rwb, rrb, av);

    // attn_out = av @ Wo^T  (B transposed: contiguous axis is k)
    gemm(av, SD, Wo_l, SD, 1, ao, SD, ROWS, SD, SD, nullptr, 0);
    add_ln_kernel<<<ROWS, 128, 0, stream>>>(ao, h, ln_attn_w + (size_t)l * SD,
                                            ln_attn_b + (size_t)l * SD, h);

    // FFN in 2048-row chunks
    for (int c = 0; c < ROWS / 2048; ++c) {
      float* hc = h + (size_t)c * 2048 * SD;
      gemm(hc, SD, W1_l, DINNER, 0, t1, DINNER, 2048, DINNER, SD, b1_l, 2);
      gemm(t1, DINNER, W2_l, SD, 0, t2, SD, 2048, SD, DINNER, b2_l, 1);
      add_ln_kernel<<<2048, 128, 0, stream>>>(t2, hc, ln_ff_w + (size_t)l * SD,
                                              ln_ff_b + (size_t)l * SD, hc);
    }
  }

  transpose_out_kernel<<<ROWS, 256, 0, stream>>>(h, (float*)d_out);
}
